// FusedSemanticLoss_69372311765270
// MI455X (gfx1250) — compile-verified, matrix operand order checked by reading, not run on hardware
//
#include <hip/hip_runtime.h>
#include <math.h>

// MI455X / gfx1250 (wave32). Bilinear 4x mask upsample done as two
// v_wmma_f32_16x16x32_f16 per mask per 16x16 output tile; semantic label map
// built with LDS atomicMax; CE gather prefetched via global_prefetch_b8.

typedef __attribute__((ext_vector_type(16))) _Float16 v16h;
typedef __attribute__((ext_vector_type(8)))  float    v8f;

#define CE_EPS 1e-7f

__global__ __launch_bounds__(128)
void FusedSemanticLoss_tile_kernel(const int*   __restrict__ y_true,     // (8,64,1)
                                   const float* __restrict__ mask_true,  // (8,64,64,64,1)
                                   const float* __restrict__ pred,       // (8,256,256,81)
                                   float*       __restrict__ partial) {  // (2048,)
  __shared__ int sem[256];                       // semantic label map for this tile
  __shared__ __align__(32) _Float16 Tbuf[4][256]; // per-wave transpose staging
  __shared__ float red[128];

  const int t     = threadIdx.x;
  const int lane  = t & 31;           // wave32
  const int wv    = t >> 5;           // 4 waves per block
  const int laneHi = lane >> 4;       // 0 or 1

  const int bx = blockIdx.x;          // 2048 = 8 * 16 * 16
  const int n  = bx >> 8;
  const int ti = (bx >> 4) & 15;
  const int tj = bx & 15;
  const int I0 = ti << 4;
  const int J0 = tj << 4;

  sem[t] = 0;
  sem[t + 128] = 0;
  __syncthreads();

  // Prefetch this tile's semantic_pred region (16 rows x 16 cols x 81 ch) so
  // the scattered CE gather is resident by the time the GEMMs finish.
  {
    const char* rowbase = (const char*)(pred +
        (((size_t)(n * 256 + I0 + (t >> 3)) * 256 + J0) * 81));
    const char* p8 = rowbase + (size_t)(t & 7) * 768;
#pragma unroll
    for (int q = 0; q < 6; ++q) __builtin_prefetch(p8 + q * 128, 0, 0);
  }

  // Upsample weight operand (A matrix, 16x32 f16; nonzero only K=0..5).
  // Half-pixel centers: y = 0.25*(I0+M) - 0.375 ; base input row a0 = 4*ti - 1,
  // so the relative weights are tile-invariant and shared by row & col passes.
  v16h aW = {};
  if (lane < 16) {
    const float g  = 0.25f * (float)lane - 0.375f;
    const float gf = floorf(g);
    const int   ci = (int)gf + 1;       // 0..4
    const float f  = g - gf;
#pragma unroll
    for (int k = 0; k < 6; ++k) {
      float w = (k == ci) ? (1.0f - f) : ((k == ci + 1) ? f : 0.0f);
      aW[k] = (_Float16)w;
    }
  }

  int acc[8];
#pragma unroll
  for (int v = 0; v < 8; ++v) acc[v] = 0;

  // Each wave owns 16 of the 64 instances.
  for (int m = 0; m < 16; ++m) {
    const int p     = (wv << 4) + m;
    const int label = y_true[(n << 6) + p];

    // B1: 32x16 f16, rows K=0..5 = clamped mask rows, cols N=0..5 = clamped
    // mask cols (edge clamp == jax's renormalized triangle kernel at borders).
    v16h b1 = {};
    if (lane < 6) {
      const int gcol = min(max(4 * tj - 1 + lane, 0), 63);
      const float* mp = mask_true + (((size_t)((n << 6) + p)) << 12) + gcol;
#pragma unroll
      for (int k = 0; k < 6; ++k) {
        const int grow = min(max(4 * ti - 1 + k, 0), 63);
        b1[k] = (_Float16)mp[(size_t)grow << 6];
      }
    }

    v8f cz = {};
    // Row upsample: T[i][c] = sum_k Wy[i][k] * M[k][c]
    v8f d1 = __builtin_amdgcn_wmma_f32_16x16x32_f16(
        false, aW, false, b1, (short)0, cz, false, false);

    // D-layout -> B-layout transpose through per-wave LDS (row-major 16x16 f16).
#pragma unroll
    for (int v = 0; v < 8; ++v) {
      const int i = v + (laneHi << 3);   // M index of D
      const int c = lane & 15;           // N index of D
      Tbuf[wv][(i << 4) + c] = (_Float16)d1[v];
    }
    v16h b2 = {};
    if (lane < 16) b2 = *(const v16h*)&Tbuf[wv][lane << 4]; // row i of T, 32B

    // Col upsample (transposed): D2[M=j][N=i] = sum_c Wx[j][c] * T[i][c]
    v8f d2 = __builtin_amdgcn_wmma_f32_16x16x32_f16(
        false, aW, false, b2, (short)0, cz, false, false);

    // Binarize (round-half-even like jnp.round) and take label max.
#pragma unroll
    for (int v = 0; v < 8; ++v) {
      int b = (int)rintf(d2[v]);
      b = min(max(b, 0), 1);
      acc[v] = max(acc[v], label * b);
    }
  }

  // Merge the 4 waves' instance maxima into the shared semantic map.
#pragma unroll
  for (int v = 0; v < 8; ++v) {
    const int i = lane & 15;
    const int j = v + (laneHi << 3);
    atomicMax(&sem[(i << 4) + j], acc[v]);
  }
  __syncthreads();

  // Categorical CE against predicted probabilities; 2 pixels per thread.
  float s = 0.0f;
#pragma unroll
  for (int q = 0; q < 2; ++q) {
    const int pix = t + (q << 7);
    const int i = pix >> 4;
    const int j = pix & 15;
    const int lab = sem[pix];
    const float pv = pred[((size_t)(n * 256 + I0 + i) * 256 + (J0 + j)) * 81 + lab];
    s += -logf(fminf(fmaxf(pv, CE_EPS), 1.0f));
  }
  red[t] = s;
  __syncthreads();
#pragma unroll
  for (int o = 64; o > 0; o >>= 1) {
    if (t < o) red[t] += red[t + o];
    __syncthreads();
  }
  if (t == 0) partial[bx] = red[0];
}

__global__ __launch_bounds__(256)
void FusedSemanticLoss_finalize_kernel(const float* __restrict__ partial,
                                       float* __restrict__ out) {
  __shared__ float red[256];
  const int t = threadIdx.x;
  float s = 0.0f;
#pragma unroll
  for (int q = 0; q < 8; ++q) s += partial[t + (q << 8)];
  red[t] = s;
  __syncthreads();
#pragma unroll
  for (int o = 128; o > 0; o >>= 1) {
    if (t < o) red[t] += red[t + o];
    __syncthreads();
  }
  if (t == 0) {
    float loss = red[0] / 524288.0f;   // mean over N*H*W = 8*256*256
    if (isnan(loss) || isinf(loss)) loss = 0.0f;
    out[0] = loss;
  }
}

extern "C" void kernel_launch(void* const* d_in, const int* in_sizes, int n_in,
                              void* d_out, int out_size, void* d_ws, size_t ws_size,
                              hipStream_t stream) {
  const int*   y_true = (const int*)d_in[0];     // (8,64,1) int32
  const float* mask   = (const float*)d_in[1];   // (8,64,64,64,1) f32
  const float* pred   = (const float*)d_in[2];   // (8,256,256,81) f32
  float* partial = (float*)d_ws;                 // 2048 floats of scratch

  FusedSemanticLoss_tile_kernel<<<2048, 128, 0, stream>>>(y_true, mask, pred, partial);
  FusedSemanticLoss_finalize_kernel<<<1, 256, 0, stream>>>(partial, (float*)d_out);
}
